// LSTM_30769145709035
// MI455X (gfx1250) — matrix operand order repaired and checked by reading, not compiled, not run on hardware
//
#include <hip/hip_runtime.h>

// ---------------------------------------------------------------------------
// 2-layer LSTM (B=64, T=512, IN=512, H=1024, OUT=1) for gfx1250 (MI455X).
// Matrix work runs on V_WMMA_F32_16X16X32_BF16 (bf16 in, fp32 accumulate).
// A-tiles are staged into LDS with GLOBAL_LOAD_ASYNC_TO_LDS_B128 (ASYNCcnt);
// B fragments are double-buffered from global with a 2x-unrolled pipeline
// (buffers alternate roles -> no register copies, no WAR hazard nops).
// ---------------------------------------------------------------------------

typedef __attribute__((ext_vector_type(16))) __bf16 v16bf;
typedef __attribute__((ext_vector_type(8)))  __bf16 v8bf;
typedef __attribute__((ext_vector_type(8)))  float  v8f;

#define LSTM_B   64
#define LSTM_T   512
#define LSTM_IN  512
#define LSTM_H   1024
#define LSTM_G   (4 * LSTM_H)   // 4096

#define GEMM_KMAX   1024        // max K across all GEMMs
#define GEMM_APAD   8           // bf16 row padding to spread LDS banks
#define GEMM_BLOCKN 512         // C columns per workgroup (8 waves x 64)

// ---------------------------------------------------------------------------
// fp32 -> bf16 conversion
// ---------------------------------------------------------------------------
__global__ void f32_to_bf16_kernel(const float* __restrict__ src,
                                   __bf16* __restrict__ dst, int n) {
  int i = blockIdx.x * blockDim.x + threadIdx.x;
  if (i < n) dst[i] = (__bf16)src[i];
}

// bias_comb = b_ih + b_hh
__global__ void bias_combine_kernel(const float* __restrict__ a,
                                    const float* __restrict__ b,
                                    float* __restrict__ o, int n) {
  int i = blockIdx.x * blockDim.x + threadIdx.x;
  if (i < n) o[i] = a[i] + b[i];
}

__global__ void zero_u32_kernel(unsigned int* __restrict__ p, int n) {
  int i = blockIdx.x * blockDim.x + threadIdx.x;
  if (i < n) p[i] = 0u;
}

// ---------------------------------------------------------------------------
// WMMA GEMM:  C[M,N] (+)= A[M,K] * W[N,K]^T  (+ bias[N])
//   A : bf16 row-major, leading dim lda
//   W : bf16 row-major [N,K] (so B-matrix column n == W row n, contiguous K)
//   C : fp32 row-major, leading dim ldc
//
// Workgroup = 8 waves, one 16x512 C tile. The 16xK A tile is shared by all
// 8 waves: staged once into LDS via async global->LDS loads, then each wave
// reads its A fragments with ds_load_b128 while B streams from global (L2-
// resident weights). The K loop is software-pipelined and unrolled by 2 so
// the two A/B buffer sets alternate producer/consumer roles: loads for the
// next K-chunk are outstanding across the current chunk's WMMAs, and no
// buffer-rotation copies are needed.
//
// Fragment packing for the 16-bit A(16x32)/B(32x16) operands (same contiguous K order in A and B):
//   A: lanes 0-15 hold K {k..k+15}, lanes 16-31 hold K {k+16..k+31}
//   B: lane n%16 holds 16 contiguous K values, half selected by lane/16
//   C: lane = column, VGPR r = row (+8 for lanes 16-31)
// ---------------------------------------------------------------------------
__launch_bounds__(256)
__global__ void wmma_gemm_bf16_kernel(const __bf16* __restrict__ A, int lda,
                                      const __bf16* __restrict__ W, int ldw,
                                      float* __restrict__ C, int ldc,
                                      const float* __restrict__ bias,
                                      int M, int N, int K, int accumulate) {
  __shared__ __bf16 smA[16][GEMM_KMAX + GEMM_APAD];

  const int lane  = threadIdx.x & 31;
  const int wavew = threadIdx.x >> 5;       // wave within block: 0..7
  const int numNB = N / GEMM_BLOCKN;        // blocks along N
  const int tm    = (blockIdx.x / numNB) << 4;            // 16 C rows
  const int tn    = (blockIdx.x % numNB) * GEMM_BLOCKN + wavew * 64;
  if (tm >= M) return;                      // block-uniform

  // ---- stage the 16xK A tile into LDS (async global->LDS, ASYNCcnt) ----
  {
    const int kch = K >> 3;                 // 16-byte chunks per row
    const int chunks = 16 * kch;
    for (int ch = threadIdx.x; ch < chunks; ch += blockDim.x) {
      const int row = ch / kch;
      const int c8  = (ch - row * kch) << 3;          // bf16 column
      const __bf16* g = A + (size_t)(tm + row) * lda + c8;
      unsigned lds = (unsigned)(size_t)&smA[row][c8]; // AS(3) byte offset
      asm volatile("global_load_async_to_lds_b128 %0, %1, off"
                   :: "v"(lds), "v"((unsigned long long)(size_t)g)
                   : "memory");
    }
    asm volatile("s_wait_asynccnt 0" ::: "memory");
  }
  __syncthreads();

  const int half = lane >> 4;               // 0: low K half, 1: high K half
  const int l16  = lane & 15;
  const int rowb = tm + half * 8;           // C row base for this lane

  v8f acc[4];
  if (accumulate) {
#pragma unroll
    for (int j = 0; j < 4; ++j) {
      const int n = tn + j * 16 + l16;
#pragma unroll
      for (int r = 0; r < 8; ++r)
        acc[j][r] = C[(size_t)(rowb + r) * ldc + n];
    }
  } else {
#pragma unroll
    for (int j = 0; j < 4; ++j)
#pragma unroll
      for (int r = 0; r < 8; ++r)
        acc[j][r] = 0.0f;
  }

  // Per-lane base pointers (K advances via immediate offsets).
  const __bf16* __restrict__ arow = &smA[l16][half * 16];  // LDS A row
  const __bf16* wptr[4];
#pragma unroll
  for (int j = 0; j < 4; ++j)
    wptr[j] = W + (size_t)(tn + j * 16 + l16) * ldw + half * 16;

  auto load_a = [&](int k) {
    v8bf a0 = *(const v8bf*)(arow + k);
    v8bf a1 = *(const v8bf*)(arow + k + 8);
    v16bf a;
#pragma unroll
    for (int i = 0; i < 8; ++i) { a[i] = a0[i]; a[8 + i] = a1[i]; }
    return a;
  };

  // ---- software-pipelined K loop, unrolled by 2 (K % 64 == 0 here) ----
  v16bf a0buf, a1buf, b0buf[4], b1buf[4];
  a0buf = load_a(0);
#pragma unroll
  for (int j = 0; j < 4; ++j) b0buf[j] = *(const v16bf*)(wptr[j]);

  int k = 0;
  for (; k < K - 64; k += 64) {
    // stage 1: prefetch k+32, compute k
#pragma unroll
    for (int j = 0; j < 4; ++j) b1buf[j] = *(const v16bf*)(wptr[j] + k + 32);
    a1buf = load_a(k + 32);
#pragma unroll
    for (int j = 0; j < 4; ++j)
      acc[j] = __builtin_amdgcn_wmma_f32_16x16x32_bf16(
          false, a0buf, false, b0buf[j], (short)0, acc[j], false, false);

    // stage 2: prefetch k+64, compute k+32
#pragma unroll
    for (int j = 0; j < 4; ++j) b0buf[j] = *(const v16bf*)(wptr[j] + k + 64);
    a0buf = load_a(k + 64);
#pragma unroll
    for (int j = 0; j < 4; ++j)
      acc[j] = __builtin_amdgcn_wmma_f32_16x16x32_bf16(
          false, a1buf, false, b1buf[j], (short)0, acc[j], false, false);
  }
  // epilogue: last two K chunks (k == K-64)
#pragma unroll
  for (int j = 0; j < 4; ++j) b1buf[j] = *(const v16bf*)(wptr[j] + k + 32);
  a1buf = load_a(k + 32);
#pragma unroll
  for (int j = 0; j < 4; ++j)
    acc[j] = __builtin_amdgcn_wmma_f32_16x16x32_bf16(
        false, a0buf, false, b0buf[j], (short)0, acc[j], false, false);
#pragma unroll
  for (int j = 0; j < 4; ++j)
    acc[j] = __builtin_amdgcn_wmma_f32_16x16x32_bf16(
        false, a1buf, false, b1buf[j], (short)0, acc[j], false, false);

#pragma unroll
  for (int j = 0; j < 4; ++j) {
    const int n = tn + j * 16 + l16;
    const float bv = bias ? bias[n] : 0.0f;
#pragma unroll
    for (int r = 0; r < 8; ++r)
      C[(size_t)(rowb + r) * ldc + n] = acc[j][r] + bv;
  }
}

// ---------------------------------------------------------------------------
// Fused LSTM cell pointwise update:
//   gates [B,4H] fp32 (i|f|g|o), c [B,H] fp32 in/out, h -> bf16 (next GEMM A)
// ---------------------------------------------------------------------------
__global__ void lstm_cell_kernel(const float* __restrict__ gates,
                                 float* __restrict__ c,
                                 __bf16* __restrict__ hbf,
                                 int Hn, int total) {
  int idx = blockIdx.x * blockDim.x + threadIdx.x;
  if (idx >= total) return;
  const int b = idx / Hn;
  const int j = idx - b * Hn;
  const float* g = gates + (size_t)b * (4 * Hn);
  const float gi = g[j];
  const float gf = g[Hn + j];
  const float gg = g[2 * Hn + j];
  const float go = g[3 * Hn + j];
  const float si = 1.0f / (1.0f + __expf(-gi));
  const float sf = 1.0f / (1.0f + __expf(-gf));
  const float so = 1.0f / (1.0f + __expf(-go));
  const float cn = sf * c[idx] + si * tanhf(gg);
  const float hn = so * tanhf(cn);
  c[idx]   = cn;
  hbf[idx] = (__bf16)hn;
}

// ---------------------------------------------------------------------------
// Per-step OUT=1 projection: out[b*T + t] = dot(h2_t[b,:], w_out) + b_out
// One block per batch element, LDS tree reduction.
// ---------------------------------------------------------------------------
__global__ void proj_kernel(const __bf16* __restrict__ h,
                            const float* __restrict__ w,
                            const float* __restrict__ bout,
                            float* __restrict__ out, int t, int Tn, int Hn) {
  __shared__ float red[256];
  const int b = blockIdx.x;
  float s = 0.0f;
  for (int j = threadIdx.x; j < Hn; j += blockDim.x)
    s += (float)h[(size_t)b * Hn + j] * w[j];
  red[threadIdx.x] = s;
  __syncthreads();
  for (int off = 128; off > 0; off >>= 1) {
    if ((int)threadIdx.x < off) red[threadIdx.x] += red[threadIdx.x + off];
    __syncthreads();
  }
  if (threadIdx.x == 0) out[(size_t)b * Tn + t] = red[0] + bout[0];
}

// ---------------------------------------------------------------------------
// Host-side orchestration
// ---------------------------------------------------------------------------
static inline void launch_gemm(const __bf16* A, int lda, const __bf16* W, int ldw,
                               float* C, int ldc, const float* bias,
                               int M, int N, int K, int accumulate,
                               hipStream_t stream) {
  const int blocks = (M / 16) * (N / GEMM_BLOCKN);
  wmma_gemm_bf16_kernel<<<blocks, 256, 0, stream>>>(A, lda, W, ldw, C, ldc,
                                                    bias, M, N, K, accumulate);
}

extern "C" void kernel_launch(void* const* d_in, const int* in_sizes, int n_in,
                              void* d_out, int out_size, void* d_ws, size_t ws_size,
                              hipStream_t stream) {
  (void)in_sizes; (void)n_in; (void)out_size; (void)ws_size;

  const float* x     = (const float*)d_in[0];   // [B,T,IN]
  const float* w_ih0 = (const float*)d_in[1];   // [4H,IN]
  const float* w_hh0 = (const float*)d_in[2];   // [4H,H]
  const float* b_ih0 = (const float*)d_in[3];
  const float* b_hh0 = (const float*)d_in[4];
  const float* w_ih1 = (const float*)d_in[5];   // [4H,H]
  const float* w_hh1 = (const float*)d_in[6];   // [4H,H]
  const float* b_ih1 = (const float*)d_in[7];
  const float* b_hh1 = (const float*)d_in[8];
  const float* w_out = (const float*)d_in[9];   // [1,H]
  const float* b_out = (const float*)d_in[10];  // [1]
  float* out = (float*)d_out;                   // [B,T,1]

  // ---- workspace carve-up (~65 MB) ----
  char* ws = (char*)d_ws;
  size_t off = 0;
  auto take = [&](size_t bytes) {
    char* p = ws + off;
    off += (bytes + 255) & ~(size_t)255;
    return p;
  };
  __bf16* xbf    = (__bf16*)take((size_t)LSTM_B * LSTM_T * LSTM_IN * 2);
  __bf16* wih0bf = (__bf16*)take((size_t)LSTM_G * LSTM_IN * 2);
  __bf16* whh0bf = (__bf16*)take((size_t)LSTM_G * LSTM_H * 2);
  __bf16* wih1bf = (__bf16*)take((size_t)LSTM_G * LSTM_H * 2);
  __bf16* whh1bf = (__bf16*)take((size_t)LSTM_G * LSTM_H * 2);
  float*  bias0  = (float*)take((size_t)LSTM_G * 4);
  float*  bias1  = (float*)take((size_t)LSTM_G * 4);
  float*  gates  = (float*)take((size_t)LSTM_B * LSTM_G * 4);
  float*  c0     = (float*)take((size_t)LSTM_B * LSTM_H * 4);
  float*  c1     = (float*)take((size_t)LSTM_B * LSTM_H * 4);
  __bf16* h0bf   = (__bf16*)take((size_t)LSTM_B * LSTM_H * 2);  // layer0 h_t
  __bf16* h1bf   = (__bf16*)take((size_t)LSTM_B * LSTM_H * 2);  // layer1 h_t

  auto cdiv = [](int a, int b) { return (a + b - 1) / b; };

  // ---- one-time conversions ----
  {
    int n = LSTM_B * LSTM_T * LSTM_IN;
    f32_to_bf16_kernel<<<cdiv(n, 256), 256, 0, stream>>>(x, xbf, n);
    n = LSTM_G * LSTM_IN;
    f32_to_bf16_kernel<<<cdiv(n, 256), 256, 0, stream>>>(w_ih0, wih0bf, n);
    n = LSTM_G * LSTM_H;
    f32_to_bf16_kernel<<<cdiv(n, 256), 256, 0, stream>>>(w_hh0, whh0bf, n);
    f32_to_bf16_kernel<<<cdiv(n, 256), 256, 0, stream>>>(w_ih1, wih1bf, n);
    f32_to_bf16_kernel<<<cdiv(n, 256), 256, 0, stream>>>(w_hh1, whh1bf, n);
    bias_combine_kernel<<<cdiv(LSTM_G, 256), 256, 0, stream>>>(b_ih0, b_hh0, bias0, LSTM_G);
    bias_combine_kernel<<<cdiv(LSTM_G, 256), 256, 0, stream>>>(b_ih1, b_hh1, bias1, LSTM_G);
  }

  // ---- zero initial state (fresh every call: harness does not re-poison) ----
  {
    int nc = LSTM_B * LSTM_H;                 // fp32 words
    zero_u32_kernel<<<cdiv(nc, 256), 256, 0, stream>>>((unsigned*)c0, nc);
    zero_u32_kernel<<<cdiv(nc, 256), 256, 0, stream>>>((unsigned*)c1, nc);
    int nh = LSTM_B * LSTM_H / 2;             // bf16 pairs as u32
    zero_u32_kernel<<<cdiv(nh, 256), 256, 0, stream>>>((unsigned*)h0bf, nh);
    zero_u32_kernel<<<cdiv(nh, 256), 256, 0, stream>>>((unsigned*)h1bf, nh);
  }

  const int total = LSTM_B * LSTM_H;          // pointwise work items
  const int pwBlocks = cdiv(total, 256);

  // ---- sequential scan over T, both layers interleaved per step ----
  for (int t = 0; t < LSTM_T; ++t) {
    // Layer 0: gates = x_t @ W_ih0^T + (b_ih0+b_hh0)
    //   A rows are x[b, t, :] -> base xbf + t*IN, lda = T*IN
    launch_gemm(xbf + (size_t)t * LSTM_IN, LSTM_T * LSTM_IN,
                wih0bf, LSTM_IN, gates, LSTM_G, bias0,
                LSTM_B, LSTM_G, LSTM_IN, /*acc=*/0, stream);
    // gates += h0_{t-1} @ W_hh0^T
    launch_gemm(h0bf, LSTM_H, whh0bf, LSTM_H, gates, LSTM_G, nullptr,
                LSTM_B, LSTM_G, LSTM_H, /*acc=*/1, stream);
    // c0,h0 update
    lstm_cell_kernel<<<pwBlocks, 256, 0, stream>>>(gates, c0, h0bf, LSTM_H, total);

    // Layer 1: gates = h0_t @ W_ih1^T + (b_ih1+b_hh1)
    launch_gemm(h0bf, LSTM_H, wih1bf, LSTM_H, gates, LSTM_G, bias1,
                LSTM_B, LSTM_G, LSTM_H, /*acc=*/0, stream);
    // gates += h1_{t-1} @ W_hh1^T
    launch_gemm(h1bf, LSTM_H, whh1bf, LSTM_H, gates, LSTM_G, nullptr,
                LSTM_B, LSTM_G, LSTM_H, /*acc=*/1, stream);
    // c1,h1 update
    lstm_cell_kernel<<<pwBlocks, 256, 0, stream>>>(gates, c1, h1bf, LSTM_H, total);

    // out[:, t] = h1_t @ w_out + b_out
    proj_kernel<<<LSTM_B, 256, 0, stream>>>(h1bf, w_out, b_out, out, t,
                                            LSTM_T, LSTM_H);
  }
}
